// GCNClassifier_81166291959946
// MI455X (gfx1250) — matrix operand order repaired and checked by reading, not compiled, not run on hardware
//
#include <hip/hip_runtime.h>
#include <hip/hip_bf16.h>
#include <stdint.h>

// ---------------------------------------------------------------------------
// GCN classifier pipeline for MI455X (gfx1250, wave32).
//   h1 = dropout(prelu(GCNagg(x @ W1) + b1))
//   h2 = dropout(prelu(GCNagg(h1 @ W2) + b2))
//   out = h2 @ Wfc + bfc
// GEMMs: v_wmma_f32_16x16x32_bf16, K fully unrolled (4 WMMAs/tile).
// A-tile staging: Tensor Data Mover (tensor_load_to_lds + s_wait_tensorcnt)
//   when available (1-D contiguous 8KB tile), LDS cooperative load otherwise.
// Graph aggregation: native global_atomic_add_f32; both 51MB feature buffers
//   are resident in the 192MB L2, plus global_prefetch_b8 lookahead on gathers.
// ---------------------------------------------------------------------------

typedef __attribute__((ext_vector_type(16))) __bf16       v16bf;
typedef __attribute__((ext_vector_type(8)))  float        v8f;
typedef __attribute__((ext_vector_type(4)))  unsigned int u32x4;
typedef __attribute__((ext_vector_type(8)))  int          i32x8;
typedef __attribute__((ext_vector_type(4)))  int          i32x4;

// TDM path: ROCm 7.2 (clang-22) 5-arg builtin. The therock TDM header implies
// the 6-arg toolchain -> fall back to cooperative loads there.
#if defined(__HIP_DEVICE_COMPILE__) && !__has_include(<hip/amd_detail/amd_gfx1250_TDM.h>) && __has_builtin(__builtin_amdgcn_tensor_load_to_lds)
#define USE_TDM 1
#else
#define USE_TDM 0
#endif

// ---------------- small utility kernels ----------------

__global__ void k_fill(float* __restrict__ p, float v, int n) {
    int i = blockIdx.x * blockDim.x + threadIdx.x;
    if (i < n) p[i] = v;
}

__global__ void k_deg_accum(const long long* __restrict__ dst,
                            float* __restrict__ deg, int E) {
    int e = blockIdx.x * blockDim.x + threadIdx.x;
    if (e < E) unsafeAtomicAdd(&deg[(size_t)dst[e]], 1.0f);
}

__global__ void k_rsqrt_inplace(float* __restrict__ p, int n) {
    int i = blockIdx.x * blockDim.x + threadIdx.x;
    if (i < n) {
        float d = p[i];
        p[i] = (d > 0.0f) ? rsqrtf(d) : 0.0f;
    }
}

// ---------------- WMMA GEMM: C[16-row tile] = A @ W ----------------
// A: n x K (f32, row major), W: K x NC (f32, row major), C: n x NC.
// One block = one 16-row tile; blockDim = (32, NC/16); wave y owns a 16-col tile.
// A tile staged into LDS via TDM DMA (contiguous 16*K floats); fragments
// converted to bf16. Layouts per CDNA5 ISA §7.12.2 (wave32).

template<int NC, int K>
__launch_bounds__(32 * (NC / 16))
__global__ void k_wmma_gemm(const float* __restrict__ A,
                            const float* __restrict__ W,
                            float* __restrict__ C) {
    __shared__ float sA[16 * K];
    const int lane    = threadIdx.x;            // 0..31 (wave32)
    const int waveCol = threadIdx.y;            // column tile index
    const int rowBase = blockIdx.x * 16;

#if USE_TDM
    if (waveCol == 0) {
        // Build D# for a 1-D contiguous copy of 16*K f32 (tile rows contiguous).
        unsigned long long ga =
            (unsigned long long)(uintptr_t)(A + (size_t)rowBase * K);
        unsigned lds = (unsigned)(uintptr_t)(&sA[0]);   // low 32b = LDS offset
        const unsigned elems = 16u * (unsigned)K;       // 2048 (fits tile_dim0)
        u32x4 g0;
        g0[0] = 1u;                                     // count=1, user mode
        g0[1] = lds;                                    // lds_addr
        g0[2] = (unsigned)(ga & 0xFFFFFFFFull);         // global_addr[31:0]
        g0[3] = (unsigned)((ga >> 32) & 0x1FFFFFFull)   // global_addr[56:32]
              | (2u << 30);                             // type=2 ("image")
        i32x8 g1 = {};
        g1[0] = (int)(2u << 16);                        // data_size=4B
        g1[1] = (int)((elems & 0xFFFFu) << 16);         // tensor_dim0[15:0]
        g1[2] = (int)((elems >> 16) & 0xFFFFu)          // tensor_dim0[31:16]
              | (1 << 16);                              // tensor_dim1 = 1
        g1[3] = (int)((elems & 0xFFFFu) << 16);         // tile_dim0 = elems
        g1[5] = (int)elems;                             // dim0 stride (unused, 1-D)
        i32x4 gz = {};
        __builtin_amdgcn_tensor_load_to_lds(g0, g1, gz, gz, 0);
        __builtin_amdgcn_s_wait_tensorcnt(0);
    }
    __syncthreads();
#else
    {
        const int nthr = 32 * (NC / 16);
        const int tid  = waveCol * 32 + lane;
        for (int i = tid; i < 16 * K; i += nthr)
            sA[i] = A[(size_t)rowBase * K + i];
        __syncthreads();
    }
#endif

    const int  mn = lane & 15;                  // row for A-frag, col for B/C-frag
    const bool hi = (lane >= 16);

    v8f acc = {};                               // f32 accumulator (8 VGPRs)

#pragma unroll
    for (int kk = 0; kk < K; kk += 32) {
        // A-fragment (16x32 bf16): lanes 0-15 hold K = e<8 ? e : e+8,
        //                          lanes 16-31 hold K = e<8 ? e+8 : e+16.
        v16bf a;
#pragma unroll
        for (int e = 0; e < 16; ++e) {
            int ka = hi ? ((e < 8) ? e + 8 : e + 16)
                        : ((e < 8) ? e     : e + 8);
            a[e] = (__bf16)sA[mn * K + kk + ka];
        }
        // B-fragment (32x16 bf16): same K order as A (lanes 0-15 K = e<8 ? e : e+8,
        //                          lanes 16-31 K = e<8 ? e+8 : e+16).
        v16bf b;
        const int kb = hi ? 8 : 0;
#pragma unroll
        for (int e = 0; e < 16; ++e)
            b[e] = (__bf16)W[(size_t)(kk + kb + e + ((e < 8) ? 0 : 8)) * NC + waveCol * 16 + mn];

        acc = __builtin_amdgcn_wmma_f32_16x16x32_bf16(
                  false, a, false, b, (short)0, acc, false, false);
    }

    // C/D layout: VGPR j holds row M = j (+8 for lanes 16-31), col = lane&15.
#pragma unroll
    for (int j = 0; j < 8; ++j) {
        int mm = j + (hi ? 8 : 0);
        C[(size_t)(rowBase + mm) * NC + waveCol * 16 + mn] = acc[j];
    }
}

// ---------------- self-loop init: agg[i] = feat[i] * dis[node]^2 ----------------
// Writes every element -> doubles as the accumulator zero-init.

template<int C>
__global__ void k_selfloop(const float* __restrict__ feat,
                           const float* __restrict__ dis,
                           float* __restrict__ agg, long long total) {
    long long i = (long long)blockIdx.x * blockDim.x + threadIdx.x;
    if (i >= total) return;
    float w = dis[i / C];
    agg[i] = feat[i] * w * w;
}

// ---------------- edge scatter: agg[dst] += feat[src] * dis[src]*dis[dst] -------
// C/4 threads per edge, float4 payload, native f32 atomics, prefetch lookahead.

template<int C>
__global__ void k_edge_scatter(const float* __restrict__ feat,
                               float* __restrict__ agg,
                               const long long* __restrict__ srcI,
                               const long long* __restrict__ dstI,
                               const float* __restrict__ dis, int E) {
    const int TPE = C / 4;
    long long tid = (long long)blockIdx.x * blockDim.x + threadIdx.x;
    long long e   = tid / TPE;
    int       sub = (int)(tid % TPE);
    if (e >= E) return;

    // Pull the gather row 8 edges ahead toward L2/L0 (global_prefetch_b8).
    if (sub == 0 && e + 8 < E) {
        long long sp = srcI[e + 8];
        __builtin_prefetch(feat + (size_t)sp * C, 0, 1);
    }

    long long s = srcI[e], d = dstI[e];
    float nrm = dis[s] * dis[d];
    const float4 v = *reinterpret_cast<const float4*>(feat + (size_t)s * C + sub * 4);
    float* o = agg + (size_t)d * C + sub * 4;
    unsafeAtomicAdd(o + 0, v.x * nrm);
    unsafeAtomicAdd(o + 1, v.y * nrm);
    unsafeAtomicAdd(o + 2, v.z * nrm);
    unsafeAtomicAdd(o + 3, v.w * nrm);
}

// ---------------- bias + PReLU + deterministic hash dropout (p=0.5) -------------

__device__ __forceinline__ unsigned wang_hash(unsigned s) {
    s = (s ^ 61u) ^ (s >> 16); s *= 9u; s ^= s >> 4;
    s *= 0x27d4eb2du; s ^= s >> 15; return s;
}

__global__ void k_prelu_dropout(float* __restrict__ buf,
                                const float* __restrict__ bias,
                                const float* __restrict__ alpha,
                                long long total, int C, unsigned salt) {
    long long i = (long long)blockIdx.x * blockDim.x + threadIdx.x;
    if (i >= total) return;
    float a = *alpha;
    float x = buf[i] + bias[(int)(i % C)];
    x = (x >= 0.0f) ? x : a * x;
    unsigned h = wang_hash((unsigned)i ^ salt);
    buf[i] = (h & 1u) ? 2.0f * x : 0.0f;       // keep-prob 0.5, scaled 1/(1-p)
}

// ---------------- final FC: out = h2 @ Wfc + bfc  (K=16, N=10) ------------------

__global__ void k_fc(const float* __restrict__ h, const float* __restrict__ Wfc,
                     const float* __restrict__ bfc, float* __restrict__ out, int n) {
    __shared__ float sW[160];
    __shared__ float sB[10];
    int t = threadIdx.x;
    if (t < 160) sW[t] = Wfc[t];
    if (t < 10)  sB[t] = bfc[t];
    __syncthreads();
    int i = blockIdx.x * blockDim.x + t;
    if (i >= n) return;
    const float4* hp = reinterpret_cast<const float4*>(h + (size_t)i * 16);
    float4 r0 = hp[0], r1 = hp[1], r2 = hp[2], r3 = hp[3];
    float hv[16] = { r0.x, r0.y, r0.z, r0.w, r1.x, r1.y, r1.z, r1.w,
                     r2.x, r2.y, r2.z, r2.w, r3.x, r3.y, r3.z, r3.w };
#pragma unroll
    for (int o = 0; o < 10; ++o) {
        float s = sB[o];
#pragma unroll
        for (int k = 0; k < 16; ++k) s += hv[k] * sW[k * 10 + o];
        out[(size_t)i * 10 + o] = s;
    }
}

// ---------------------------------------------------------------------------

extern "C" void kernel_launch(void* const* d_in, const int* in_sizes, int n_in,
                              void* d_out, int out_size, void* d_ws, size_t ws_size,
                              hipStream_t stream) {
    const float*      x   = (const float*)d_in[0];
    const long long*  ei  = (const long long*)d_in[1];   // int64 (2, E) row-major
    const float*      W1  = (const float*)d_in[2];
    const float*      b1  = (const float*)d_in[3];
    const float*      W2  = (const float*)d_in[4];
    const float*      b2  = (const float*)d_in[5];
    const float*      pa  = (const float*)d_in[6];
    const float*      Wfc = (const float*)d_in[7];
    const float*      bfc = (const float*)d_in[8];
    float*            out = (float*)d_out;

    const int n = in_sizes[0] / 128;       // 100000 (multiple of 16: 6250 tiles)
    const int E = in_sizes[1] / 2;         // 1600000
    const long long* src = ei;
    const long long* dst = ei + E;

    // Workspace layout (floats): dis | XW(n*128) | AG1(n*128) | HW2(n*16) | AG2(n*16)
    float* ws  = (float*)d_ws;
    float* dis = ws;
    float* XW  = dis + n;
    float* AG1 = XW  + (size_t)n * 128;
    float* HW2 = AG1 + (size_t)n * 128;
    float* AG2 = HW2 + (size_t)n * 16;

    const int B = 256;
    const unsigned gN = (n + B - 1) / B;

    // --- symmetric normalization: deg (self-loop => init 1.0) -> rsqrt ---
    k_fill<<<gN, B, 0, stream>>>(dis, 1.0f, n);
    k_deg_accum<<<(E + B - 1) / B, B, 0, stream>>>(dst, dis, E);
    k_rsqrt_inplace<<<gN, B, 0, stream>>>(dis, n);

    // --- layer 1: XW = x @ W1 ; agg = selfloop + edge scatter ; act ---
    dim3 gT(n / 16), t8(32, 8), t1(32, 1);
    k_wmma_gemm<128, 128><<<gT, t8, 0, stream>>>(x, W1, XW);
    long long tot1 = (long long)n * 128;
    k_selfloop<128><<<(unsigned)((tot1 + B - 1) / B), B, 0, stream>>>(XW, dis, AG1, tot1);
    long long st1 = (long long)E * 32;
    k_edge_scatter<128><<<(unsigned)((st1 + B - 1) / B), B, 0, stream>>>(XW, AG1, src, dst, dis, E);
    k_prelu_dropout<<<(unsigned)((tot1 + B - 1) / B), B, 0, stream>>>(AG1, b1, pa, tot1, 128, 0x9E3779B9u);

    // --- layer 2: HW2 = h1 @ W2 ; agg ; act ---
    k_wmma_gemm<16, 128><<<gT, t1, 0, stream>>>(AG1, W2, HW2);
    long long tot2 = (long long)n * 16;
    k_selfloop<16><<<(unsigned)((tot2 + B - 1) / B), B, 0, stream>>>(HW2, dis, AG2, tot2);
    long long st2 = (long long)E * 4;
    k_edge_scatter<16><<<(unsigned)((st2 + B - 1) / B), B, 0, stream>>>(HW2, AG2, src, dst, dis, E);
    k_prelu_dropout<<<(unsigned)((tot2 + B - 1) / B), B, 0, stream>>>(AG2, b2, pa, tot2, 16, 0x85EBCA6Bu);

    // --- classifier head ---
    k_fc<<<gN, B, 0, stream>>>(AG2, Wfc, bfc, out, n);
}
